// TPA_27118423507346
// MI455X (gfx1250) — matrix operand order repaired and checked by reading, not compiled, not run on hardware
//
#include <hip/hip_runtime.h>

typedef __attribute__((ext_vector_type(16))) __bf16 v16bf;
typedef __attribute__((ext_vector_type(8)))  __bf16 v8bf;
typedef __attribute__((ext_vector_type(8)))  float  v8f;
typedef unsigned short u16;

#define B_  2
#define L_  2048
#define DM  1024
#define H_  16
#define RK  256
#define RH  16
#define HD  64

__device__ __forceinline__ __bf16 f2bf(float f) {
  unsigned u = __builtin_bit_cast(unsigned, f);
  unsigned r = u + 0x7FFFu + ((u >> 16) & 1u);  // round-to-nearest-even
  u16 h = (u16)(r >> 16);
  return __builtin_bit_cast(__bf16, h);
}

// Async global->LDS copy, 16 bytes per lane, tracked by ASYNCcnt.
// Low 32 bits of a flat LDS pointer are the LDS byte offset (ISA 10.2).
__device__ __forceinline__ void async_g2l_b128(u16* lds_ptr, const u16* gptr) {
  unsigned loff = (unsigned)(size_t)lds_ptr;
  asm volatile("global_load_async_to_lds_b128 %0, %1, off"
               :: "v"(loff), "v"(gptr) : "memory");
}
__device__ __forceinline__ void wait_asynccnt0() {
  asm volatile("s_wait_asynccnt 0x0" ::: "memory");
}

// ---------------------------------------------------------------------------
// Kernel 0: one-shot f32 -> bf16 conversion of x and {Wq,Wk,Wv} so the GEMM
// inner loop is pure loads + WMMA (no per-iteration VALU rounding sequences).
// ---------------------------------------------------------------------------
__global__ void __launch_bounds__(256) to_bf16(
    const float* __restrict__ x,
    const float* __restrict__ Wq, const float* __restrict__ Wk,
    const float* __restrict__ Wv,
    u16* __restrict__ xbf, u16* __restrict__ wbf)
{
  const int idx = blockIdx.x * 256 + threadIdx.x;
  const int xn = B_ * L_ * DM;            // 4194304
  if (idx < xn) {
    xbf[idx] = __builtin_bit_cast(u16, f2bf(x[idx]));
  } else {
    int j = idx - xn;                     // 0 .. 3*RK*DM-1
    int which = j / (RK * DM);
    int off   = j - which * (RK * DM);
    const float* W = (which == 0) ? Wq : (which == 1) ? Wk : Wv;
    wbf[j] = __builtin_bit_cast(u16, f2bf(W[off]));
  }
}

// ---------------------------------------------------------------------------
// Kernel 1: low-rank projections  low = x @ W^T   (M=4096, N=256, K=1024) x3
// One wave per 16x64 output tile: A-fragment reused across 4 B-fragments,
// 4 WMMAs per K-step of 32.
// ---------------------------------------------------------------------------
__global__ void __launch_bounds__(32) proj_gemm(
    const u16* __restrict__ xbf, const u16* __restrict__ wbf,
    float* __restrict__ qlo, float* __restrict__ klo, float* __restrict__ vlo)
{
  const int lane = threadIdx.x;
  const int mt    = blockIdx.x;          // 256 M-tiles of 16
  const int nbase = blockIdx.y * 64;     // 4 N-groups of 64
  const int which = blockIdx.z;
  float* out = (which == 0) ? qlo : (which == 1) ? klo : vlo;

  const int n16   = lane & 15;
  const int hi    = lane >> 4;
  const int base0 = hi * 8;              // A: K sub-offset per lane half
  const int bhalf = hi * 16;             // B: K half per lane half

  const u16* xrow = xbf + (size_t)(mt * 16 + n16) * DM;
  const u16* wrow0 = wbf + (size_t)which * RK * DM + (size_t)(nbase + n16) * DM;
  const u16* wrow1 = wrow0 + (size_t)16 * DM;
  const u16* wrow2 = wrow0 + (size_t)32 * DM;
  const u16* wrow3 = wrow0 + (size_t)48 * DM;

  v8f acc0 = {}, acc1 = {}, acc2 = {}, acc3 = {};
  for (int kb = 0; kb < DM; kb += 32) {
    __builtin_prefetch(xrow + kb + 256, 0, 3);
    __builtin_prefetch(wrow0 + kb + 256, 0, 3);
    v16bf a;
    {
      v8bf c0 = *(const v8bf*)(xrow + kb + 2 * base0);
      v8bf c1 = *(const v8bf*)(xrow + kb + 2 * base0 + 8);
#pragma unroll
      for (int e = 0; e < 8; ++e) { a[e] = c0[e]; a[e + 8] = c1[e]; }
    }
    v16bf b0 = *(const v16bf*)(wrow0 + kb + bhalf);
    v16bf b1 = *(const v16bf*)(wrow1 + kb + bhalf);
    v16bf b2 = *(const v16bf*)(wrow2 + kb + bhalf);
    v16bf b3 = *(const v16bf*)(wrow3 + kb + bhalf);
    acc0 = __builtin_amdgcn_wmma_f32_16x16x32_bf16(false, a, false, b0, (short)0, acc0, false, false);
    acc1 = __builtin_amdgcn_wmma_f32_16x16x32_bf16(false, a, false, b1, (short)0, acc1, false, false);
    acc2 = __builtin_amdgcn_wmma_f32_16x16x32_bf16(false, a, false, b2, (short)0, acc2, false, false);
    acc3 = __builtin_amdgcn_wmma_f32_16x16x32_bf16(false, a, false, b3, (short)0, acc3, false, false);
  }
  const int row0 = mt * 16 + base0;      // C: element j -> row j+8*hi, col n16
#pragma unroll
  for (int j = 0; j < 8; ++j) {
    float* o = out + (size_t)(row0 + j) * RK + nbase + n16;
    o[0]  = acc0[j];
    o[16] = acc1[j];
    o[32] = acc2[j];
    o[48] = acc3[j];
  }
}

// ---------------------------------------------------------------------------
// Kernel 2: U-expansion (rank16 -> head_dim64) + RoPE; emit bf16 Q/K
// One wave per (b,l,h); lane owns dims (2*lane, 2*lane+1) == one RoPE pair.
// ---------------------------------------------------------------------------
__global__ void __launch_bounds__(256) expand_rope(
    const float* __restrict__ qlo, const float* __restrict__ klo,
    const float* __restrict__ U, u16* __restrict__ qbf, u16* __restrict__ kbf)
{
  const int lane = threadIdx.x & 31;
  const int wid  = (blockIdx.x * blockDim.x + threadIdx.x) >> 5;
  const int h = wid % H_;
  const int l = (wid / H_) % L_;
  const int b = wid / (H_ * L_);

  const size_t lowbase = (size_t)(b * L_ + l) * RK + h * RH;
  const float* ql = qlo + lowbase;
  const float* kl = klo + lowbase;
  const int d0 = 2 * lane;
  const float* u0 = U + (size_t)(h * HD + d0) * RH;
  const float* u1 = u0 + RH;

  float q0 = 0.f, q1 = 0.f, k0 = 0.f, k1 = 0.f;
#pragma unroll
  for (int r = 0; r < RH; ++r) {
    float qv = ql[r], kv = kl[r];
    q0 += u0[r] * qv; q1 += u1[r] * qv;
    k0 += u0[r] * kv; k1 += u1[r] * kv;
  }
  float theta = __powf(10000.f, -(float)lane / 32.f);
  float s, c;
  __sincosf((float)l * theta, &s, &c);
  float qr0 = q0 * c - q1 * s, qr1 = q0 * s + q1 * c;
  float kr0 = k0 * c - k1 * s, kr1 = k0 * s + k1 * c;

  const size_t obase = ((size_t)(b * H_ + h) * L_ + l) * HD + d0;
  unsigned qp = (unsigned)__builtin_bit_cast(u16, f2bf(qr0)) |
                ((unsigned)__builtin_bit_cast(u16, f2bf(qr1)) << 16);
  unsigned kp = (unsigned)__builtin_bit_cast(u16, f2bf(kr0)) |
                ((unsigned)__builtin_bit_cast(u16, f2bf(kr1)) << 16);
  *(unsigned*)(qbf + obase) = qp;
  *(unsigned*)(kbf + obase) = kp;
}

// ---------------------------------------------------------------------------
// Kernel 2b: v_low -> bf16, transposed to [b][h][r][L] for contiguous B-frags
// ---------------------------------------------------------------------------
__global__ void __launch_bounds__(256) v_transpose(
    const float* __restrict__ vlo, u16* __restrict__ vT)
{
  int idx = blockIdx.x * 256 + threadIdx.x;   // 2^20 elements
  int l = idx & (L_ - 1);
  int r = (idx >> 11) & 15;
  int h = (idx >> 15) & 15;
  int b = idx >> 19;
  float v = vlo[(size_t)(b * L_ + l) * RK + h * RH + r];
  vT[((size_t)(b * H_ + h) * RH + r) * L_ + l] = __builtin_bit_cast(u16, f2bf(v));
}

// ---------------------------------------------------------------------------
// Kernel 3: flash attention, 32-key tiles. One wave per 16-row Q tile,
// 4 waves / block sharing one (b,h). Per tile:
//   - K-tile (32 keys x 64 dims, 4KB) staged into LDS ONCE per block via
//     GLOBAL_LOAD_ASYNC_TO_LDS_B128 (ASYNCcnt + s_wait_asynccnt + barrier),
//     4x less L2 traffic than per-wave reloads.
//   - 4 S-WMMAs (2 col-tiles x K=64) with B-fragments read from LDS.
//   - one fused softmax pass (shuffle tree shared by both column halves).
//   - P transposed through LDS into a dense 16x32 A-fragment; 1 dense PV WMMA.
// ---------------------------------------------------------------------------
__global__ void __launch_bounds__(128) flash_attn(
    const u16* __restrict__ qbf, const u16* __restrict__ kbf,
    const u16* __restrict__ vT, const int* __restrict__ mask,
    float* __restrict__ olo)
{
  __shared__ __align__(32) u16 Ktile[32 * HD];      // 4 KB, shared by block
  __shared__ __align__(32) u16 Plds[4][16 * 32];    // 4 KB, per-wave
  const int tid  = threadIdx.x;
  const int lane = tid & 31;
  const int w    = tid >> 5;
  const int qt = blockIdx.x * 4 + w;     // 128 q-tiles
  const int h  = blockIdx.y;
  const int b  = blockIdx.z;
  const int q0 = qt * 16;

  const int n16   = lane & 15;
  const int hi    = lane >> 4;
  const int base0 = hi * 8;
  const int bhalf = hi * 16;

  const u16* qbase = qbf + (size_t)(b * H_ + h) * L_ * HD;
  const u16* kbase = kbf + (size_t)(b * H_ + h) * L_ * HD;
  const u16* vbase = vT  + (size_t)(b * H_ + h) * RH * L_;
  const int* mrow  = mask + b * L_;

  // async-stage chunk assignment: 256 16B-chunks = 32 rows x 8 chunks
  const int c0row = tid >> 3,          c0ch = (tid & 7) * 8;
  const int c1row = (tid + 128) >> 3,  c1ch = ((tid + 128) & 7) * 8;

  // Q A-fragments for K-dim 0..31 and 32..63
  const u16* qrow = qbase + (size_t)(q0 + n16) * HD;
  v16bf aq0, aq1;
  {
    v8bf c0 = *(const v8bf*)(qrow + 2 * base0);
    v8bf c1 = *(const v8bf*)(qrow + 2 * base0 + 8);
    v8bf c2 = *(const v8bf*)(qrow + 32 + 2 * base0);
    v8bf c3 = *(const v8bf*)(qrow + 40 + 2 * base0);
#pragma unroll
    for (int e = 0; e < 8; ++e) {
      aq0[e] = c0[e]; aq0[e + 8] = c1[e];
      aq1[e] = c2[e]; aq1[e + 8] = c3[e];
    }
  }

  float mrun[8], lrun[8], alpha[8];
  v8f acc = {};
#pragma unroll
  for (int j = 0; j < 8; ++j) { mrun[j] = -3.0e38f; lrun[j] = 0.f; }

  for (int kt = 0; kt < L_ / 32; ++kt) {
    const int m0 = kt * 32;
    // cooperative async fill of the K-tile (each thread: 2 x 16 bytes)
    async_g2l_b128(&Ktile[c0row * HD + c0ch],
                   kbase + (size_t)(m0 + c0row) * HD + c0ch);
    async_g2l_b128(&Ktile[c1row * HD + c1ch],
                   kbase + (size_t)(m0 + c1row) * HD + c1ch);
    wait_asynccnt0();
    __syncthreads();                      // K-tile visible to all waves

    // K B-fragments from LDS (col = tile-local key, K-dim contiguous)
    v16bf bk00 = *(const v16bf*)&Ktile[n16 * HD + bhalf];
    v16bf bk01 = *(const v16bf*)&Ktile[n16 * HD + 32 + bhalf];
    v16bf bk10 = *(const v16bf*)&Ktile[(16 + n16) * HD + bhalf];
    v16bf bk11 = *(const v16bf*)&Ktile[(16 + n16) * HD + 32 + bhalf];
    v8f s0 = {}, s1 = {};
    s0 = __builtin_amdgcn_wmma_f32_16x16x32_bf16(false, aq0, false, bk00, (short)0, s0, false, false);
    s0 = __builtin_amdgcn_wmma_f32_16x16x32_bf16(false, aq1, false, bk01, (short)0, s0, false, false);
    s1 = __builtin_amdgcn_wmma_f32_16x16x32_bf16(false, aq0, false, bk10, (short)0, s1, false, false);
    s1 = __builtin_amdgcn_wmma_f32_16x16x32_bf16(false, aq1, false, bk11, (short)0, s1, false, false);

    const bool on0 = (mrow[m0 + n16] != 0);        // lane's key col, tile 0
    const bool on1 = (mrow[m0 + 16 + n16] != 0);   // lane's key col, tile 1
#pragma unroll
    for (int j = 0; j < 8; ++j) {
      float sv0 = on0 ? s0[j] * 0.125f : -3.0e38f;   // 1/sqrt(64)
      float sv1 = on1 ? s1[j] * 0.125f : -3.0e38f;
      float rmax = fmaxf(sv0, sv1);
      rmax = fmaxf(rmax, __shfl_xor(rmax, 1, 16));
      rmax = fmaxf(rmax, __shfl_xor(rmax, 2, 16));
      rmax = fmaxf(rmax, __shfl_xor(rmax, 4, 16));
      rmax = fmaxf(rmax, __shfl_xor(rmax, 8, 16));
      float mnew = fmaxf(mrun[j], rmax);
      float a  = (mnew < -1.0e37f) ? 1.f : __expf(mrun[j] - mnew);
      float p0 = (sv0  < -1.0e37f) ? 0.f : __expf(sv0 - mnew);
      float p1 = (sv1  < -1.0e37f) ? 0.f : __expf(sv1 - mnew);
      float rsum = p0 + p1;
      rsum += __shfl_xor(rsum, 1, 16);
      rsum += __shfl_xor(rsum, 2, 16);
      rsum += __shfl_xor(rsum, 4, 16);
      rsum += __shfl_xor(rsum, 8, 16);
      mrun[j] = mnew;
      lrun[j] = lrun[j] * a + rsum;
      alpha[j] = a;
      // P: C layout (row j+base0, cols n16 and 16+n16) -> LDS row-major 16x32
      Plds[w][(j + base0) * 32 + n16]      = __builtin_bit_cast(u16, f2bf(p0));
      Plds[w][(j + base0) * 32 + 16 + n16] = __builtin_bit_cast(u16, f2bf(p1));
    }
    // barrier: orders P store->load transpose AND licenses next K-tile
    // overwrite (all Ktile ds_loads above happened before this point)
    __syncthreads();
    v16bf ap;
    {
      v8bf pr0 = *(const v8bf*)&Plds[w][n16 * 32 + 2 * base0];      // K=2*base0..+7
      v8bf pr1 = *(const v8bf*)&Plds[w][n16 * 32 + 2 * base0 + 8];  // K=2*base0+8..+15
#pragma unroll
      for (int e = 0; e < 8; ++e) { ap[e] = pr0[e]; ap[e + 8] = pr1[e]; }
    }

    // V tile B-fragment, dense: K = bhalf+e -> key m0+bhalf+e
    v16bf bv = *(const v16bf*)(vbase + (size_t)n16 * L_ + m0 + bhalf);

#pragma unroll
    for (int j = 0; j < 8; ++j) acc[j] *= alpha[j];
    acc = __builtin_amdgcn_wmma_f32_16x16x32_bf16(false, ap, false, bv,
                                                  (short)0, acc, false, false);
  }

#pragma unroll
  for (int j = 0; j < 8; ++j) {
    int seq = q0 + j + base0;
    olo[(size_t)(b * L_ + seq) * RK + h * RH + n16] =
        acc[j] / fmaxf(lrun[j], 1e-20f);
  }
}

// ---------------------------------------------------------------------------
// Kernel 4: final V projection (rank16 -> 64). Tiny (0.13 GFLOP); VALU.
// ---------------------------------------------------------------------------
__global__ void __launch_bounds__(256) out_proj(
    const float* __restrict__ olo, const float* __restrict__ V,
    float* __restrict__ out)
{
  int idx = blockIdx.x * 256 + threadIdx.x;   // B*L*DM = 4194304
  int d = idx & 63;
  int h = (idx >> 6) & 15;
  size_t bl = (size_t)(idx >> 10);
  const float* o = olo + bl * RK + h * RH;
  const float* v = V + (size_t)h * RH * HD + d;
  float s = 0.f;
#pragma unroll
  for (int r = 0; r < RH; ++r) s += o[r] * v[(size_t)r * HD];
  out[idx] = s;
}

extern "C" void kernel_launch(void* const* d_in, const int* in_sizes, int n_in,
                              void* d_out, int out_size, void* d_ws, size_t ws_size,
                              hipStream_t stream) {
  const float* x  = (const float*)d_in[0];
  const float* Wq = (const float*)d_in[1];
  const float* Wk = (const float*)d_in[2];
  const float* Wv = (const float*)d_in[3];
  const float* U  = (const float*)d_in[4];
  const float* V  = (const float*)d_in[5];
  const int* mask = (const int*)d_in[6];
  float* out = (float*)d_out;

  char* ws = (char*)d_ws;                       // ~44 MB used
  float* qlo = (float*)(ws);                    // 4 MB fp32 [B*L][256]
  float* klo = (float*)(ws + (4u  << 20));      // 4 MB
  float* vlo = (float*)(ws + (8u  << 20));      // 4 MB
  u16*   qbf = (u16*)  (ws + (12u << 20));      // 8 MB bf16 [b][h][L][64]
  u16*   kbf = (u16*)  (ws + (20u << 20));      // 8 MB
  u16*   vT  = (u16*)  (ws + (28u << 20));      // 2 MB bf16 [b][h][16][L]
  float* olo = (float*)(ws + (30u << 20));      // 4 MB fp32 [b][l][h][16]
  u16*   xbf = (u16*)  (ws + (34u << 20));      // 8 MB bf16 [B*L][1024]
  u16*   wbf = (u16*)  (ws + (42u << 20));      // 1.5 MB bf16 [3][256][1024]

  const int conv_n = B_ * L_ * DM + 3 * RK * DM;          // 4980736
  to_bf16<<<conv_n / 256, 256, 0, stream>>>(x, Wq, Wk, Wv, xbf, wbf);
  dim3 g1(256, 4, 3);
  proj_gemm<<<g1, 32, 0, stream>>>(xbf, wbf, qlo, klo, vlo);
  expand_rope<<<(B_ * L_ * H_ * 32) / 256, 256, 0, stream>>>(qlo, klo, U, qbf, kbf);
  v_transpose<<<(B_ * H_ * RH * L_) / 256, 256, 0, stream>>>(vlo, vT);
  dim3 g3(L_ / 16 / 4, H_, B_);
  flash_attn<<<g3, 128, 0, stream>>>(qbf, kbf, vT, mask, olo);
  out_proj<<<(B_ * L_ * DM) / 256, 256, 0, stream>>>(olo, V, out);
}
